// MultiScaleE3Layer_42958262895189
// MI455X (gfx1250) — matrix operand order repaired and checked by reading, not compiled, not run on hardware
//
#include <hip/hip_runtime.h>
#include <hip/hip_bf16.h>

// ---------------------------------------------------------------------------
// MultiScaleE3Layer for gfx1250 (MI455X). bf16 WMMA (16x16x32, f32 acc) for all
// GEMMs. Edge MLPs tiled 64 edges/block (4 M-tiles per wave, B fragment reused
// 4x); vectorized (float4) LDS staging; node kernel 32 nodes/block.
// ---------------------------------------------------------------------------

#define NND   8192
#define HH    128
#define RR    32
#define BDD   6
#define EE    262144
#define EBB   16384
#define NSC   6

typedef __attribute__((ext_vector_type(16))) __bf16 v16bf;
typedef __attribute__((ext_vector_type(8)))  float  v8f;

__device__ __forceinline__ v8f wmma_bf16(v16bf a, v16bf b, v8f c) {
    return __builtin_amdgcn_wmma_f32_16x16x32_bf16(false, a, false, b, (short)0, c,
                                                   false, false);
}

// A fragment (16x32 bf16, M x K) from an LDS tile, row-major with given stride.
// Layout (ISA 7.12.2): lane m = lane&15; K-offset 0 (lanes 0-15) / 8 (lanes 16-31);
// VGPR v in [0,3]: K = koff+2v,2v+1 ; VGPR v in [4,7]: K = 16+koff+2(v-4),...
__device__ __forceinline__ v16bf frag_a_lds(const __bf16* A, int stride, int kb, int lane) {
    int m    = lane & 15;
    int koff = (lane >> 4) << 3;           // 0 or 8
    const __bf16* p = A + m * stride + kb + koff;
    v16bf a;
#pragma unroll
    for (int v = 0; v < 4; ++v) { a[2*v]   = p[2*v];      a[2*v+1]   = p[2*v+1]; }
#pragma unroll
    for (int v = 0; v < 4; ++v) { a[8+2*v] = p[16+2*v];   a[8+2*v+1] = p[16+2*v+1]; }
    return a;
}

// B fragment (32x16 bf16, K x N) from a pre-transposed weight BT[N][Kpad] (bf16,
// row-major). Layout: lane n = lane&15; same K order as frag_a_lds: lanes 0-15 hold
// K=0..7,16..23, lanes 16-31 hold K=8..15,24..31.
__device__ __forceinline__ v16bf frag_b_glb(const __bf16* BT, int stride, int kb,
                                            int nbase, int lane) {
    int n  = nbase + (lane & 15);
    int k2 = (lane >> 4) << 3;             // 0 or 8
    const __bf16* p = BT + (size_t)n * stride + kb + k2;
    v16bf b;
#pragma unroll
    for (int j = 0; j < 8; ++j) { b[j] = p[j]; b[8+j] = p[16+j]; }
    return b;
}

__device__ __forceinline__ float silu_f(float v) {
    return v / (1.0f + __expf(-v));
}

// ---------------------------------------------------------------------------
// Conversion / prep kernels
// ---------------------------------------------------------------------------
__global__ void convert_x_kernel(const float* __restrict__ x, __bf16* __restrict__ xb, int n) {
    int i = blockIdx.x * 256 + threadIdx.x;
    if (i < n) xb[i] = (__bf16)x[i];
}

// src: fp32 [K][Nc] row-major  ->  dst: bf16 [Nc][Kpad] row-major (transposed, zero pad)
__global__ void transpose_convert_kernel(const float* __restrict__ src, __bf16* __restrict__ dst,
                                         int K, int Nc, int Kpad) {
    int total = Nc * Kpad;
    for (int i = blockIdx.x * 256 + threadIdx.x; i < total; i += gridDim.x * 256) {
        int n = i / Kpad;
        int k = i - n * Kpad;
        dst[i] = (k < K) ? (__bf16)src[(size_t)k * Nc + n] : (__bf16)0.0f;
    }
}

__global__ void finalize_mean_kernel(float* __restrict__ acc, const float* __restrict__ cnt,
                                     int total) {
    int i = blockIdx.x * 256 + threadIdx.x;
    if (i < total) acc[i] = acc[i] / fmaxf(cnt[i >> 7], 1.0f);
}

// ---------------------------------------------------------------------------
// Spatial-scale edge MLP: 64 edges/block, 256 threads = 8 waves, 4 M-tiles/wave.
// GEMM1: [64,288]x[288,128], GEMM2: [64,128]x[128,128]; cosine-cutoff weight;
// scatter-add (float atomics) + edge counts.
// ---------------------------------------------------------------------------
__launch_bounds__(256, 2)
__global__ void edge_mlp_spatial(const __bf16* __restrict__ xb,
                                 const float*  __restrict__ pos,
                                 const int*    __restrict__ row,
                                 const int*    __restrict__ col,
                                 const __bf16* __restrict__ W1T, const float* __restrict__ b1,
                                 const __bf16* __restrict__ W2T, const float* __restrict__ b2,
                                 float cutoff,
                                 float* __restrict__ acc, float* __restrict__ cnt) {
    constexpr int MT = 64;                 // edges per block
    constexpr int K1 = 2 * HH + RR;        // 288
    constexpr int S1 = K1 + 8;             // 296 (pad; 592 B row, 16B-aligned)
    constexpr int S2 = HH + 8;             // 136
    __shared__ __bf16 A [MT * S1];
    __shared__ __bf16 H1[MT * S2];
    __shared__ float  dsh[MT], cwsh[MT];
    __shared__ int    rowsh[MT], colsh[MT];

    const int tid  = threadIdx.x;
    const int lane = tid & 31;
    const int wave = tid >> 5;
    const int e0   = blockIdx.x * MT;

    if (tid < MT) {
        int e = e0 + tid;
        int r = row[e], c = col[e];
        rowsh[tid] = r; colsh[tid] = c;
        float dx = pos[c*3+0] - pos[r*3+0];
        float dy = pos[c*3+1] - pos[r*3+1];
        float dz = pos[c*3+2] - pos[r*3+2];
        float d  = sqrtf(dx*dx + dy*dy + dz*dz);
        dsh[tid] = d;
        cwsh[tid] = (d < cutoff) ? 0.5f * (__cosf(d * 3.14159265358979323846f / cutoff) + 1.0f)
                                 : 0.0f;
        atomicAdd(&cnt[r], 1.0f);
    }
    __syncthreads();

    // Stage x[row] | x[col] vectorized: 8 bf16 (16B) per copy, 32 chunks/edge.
    for (int idx = tid; idx < MT * 32; idx += 256) {
        int m  = idx >> 5;
        int cp = idx & 31;                 // 0..15 -> row part, 16..31 -> col part
        const float4* src;
        int f;
        if (cp < 16) { src = (const float4*)(xb + (size_t)rowsh[m] * HH) + cp;        f = cp * 8; }
        else         { src = (const float4*)(xb + (size_t)colsh[m] * HH) + (cp - 16); f = HH + (cp - 16) * 8; }
        *(float4*)(A + m * S1 + f) = *src;
    }
    // RBF tail (computed): 32 values per edge.
    for (int idx = tid; idx < MT * RR; idx += 256) {
        int m = idx >> 5;
        int j = idx & 31;
        float center = cutoff * (float)j * (1.0f / 31.0f);
        float w      = cutoff * (1.0f / 64.0f);          // c/R*0.5
        float t      = (dsh[m] - center) / w;
        A[m * S1 + 2 * HH + j] = (__bf16)__expf(-0.5f * t * t);
    }
    __syncthreads();

    const int nbase = wave * 16;
    const int ncol  = nbase + (lane & 15);
    const int mofs  = (lane >> 4) << 3;

    // GEMM1: each B fragment feeds 4 WMMAs; A fragments preloaded so DS loads
    // stay in flight (partial dscnt waits instead of full serialization).
    v8f acc1[4] = {};
#pragma unroll
    for (int kt = 0; kt < K1 / 32; ++kt) {
        v16bf b  = frag_b_glb(W1T, K1, kt * 32, nbase, lane);
        v16bf a0 = frag_a_lds(A + 0 * 16 * S1, S1, kt * 32, lane);
        v16bf a1 = frag_a_lds(A + 1 * 16 * S1, S1, kt * 32, lane);
        v16bf a2 = frag_a_lds(A + 2 * 16 * S1, S1, kt * 32, lane);
        v16bf a3 = frag_a_lds(A + 3 * 16 * S1, S1, kt * 32, lane);
        acc1[0] = wmma_bf16(a0, b, acc1[0]);
        acc1[1] = wmma_bf16(a1, b, acc1[1]);
        acc1[2] = wmma_bf16(a2, b, acc1[2]);
        acc1[3] = wmma_bf16(a3, b, acc1[3]);
    }

    float bias1 = b1[ncol];
#pragma unroll
    for (int mt = 0; mt < 4; ++mt)
#pragma unroll
        for (int r = 0; r < 8; ++r) {
            float v = silu_f(acc1[mt][r] + bias1);
            H1[(mt * 16 + r + mofs) * S2 + ncol] = (__bf16)v;
        }
    __syncthreads();

    v8f acc2[4] = {};
#pragma unroll
    for (int kt = 0; kt < HH / 32; ++kt) {
        v16bf b  = frag_b_glb(W2T, HH, kt * 32, nbase, lane);
        v16bf a0 = frag_a_lds(H1 + 0 * 16 * S2, S2, kt * 32, lane);
        v16bf a1 = frag_a_lds(H1 + 1 * 16 * S2, S2, kt * 32, lane);
        v16bf a2 = frag_a_lds(H1 + 2 * 16 * S2, S2, kt * 32, lane);
        v16bf a3 = frag_a_lds(H1 + 3 * 16 * S2, S2, kt * 32, lane);
        acc2[0] = wmma_bf16(a0, b, acc2[0]);
        acc2[1] = wmma_bf16(a1, b, acc2[1]);
        acc2[2] = wmma_bf16(a2, b, acc2[2]);
        acc2[3] = wmma_bf16(a3, b, acc2[3]);
    }

    float bias2 = b2[ncol];
#pragma unroll
    for (int mt = 0; mt < 4; ++mt)
#pragma unroll
        for (int r = 0; r < 8; ++r) {
            int   m = mt * 16 + r + mofs;
            float v = (acc2[mt][r] + bias2) * cwsh[m];
            atomicAdd(&acc[(size_t)rowsh[m] * HH + ncol], v);
        }
}

// ---------------------------------------------------------------------------
// Bond-scale edge MLP: same structure, K padded 262 -> 288 (weights zero-padded).
// ---------------------------------------------------------------------------
__launch_bounds__(256, 2)
__global__ void edge_mlp_bond(const __bf16* __restrict__ xb,
                              const int*    __restrict__ row,
                              const int*    __restrict__ col,
                              const float*  __restrict__ attr,
                              const __bf16* __restrict__ W1T, const float* __restrict__ b1,
                              const __bf16* __restrict__ W2T, const float* __restrict__ b2,
                              float* __restrict__ acc, float* __restrict__ cnt) {
    constexpr int MT = 64;
    constexpr int KR = 2 * HH + BDD;       // 262 real
    constexpr int K1 = 288;                // padded
    constexpr int S1 = K1 + 8;
    constexpr int S2 = HH + 8;
    __shared__ __bf16 A [MT * S1];
    __shared__ __bf16 H1[MT * S2];
    __shared__ int    rowsh[MT], colsh[MT];

    const int tid  = threadIdx.x;
    const int lane = tid & 31;
    const int wave = tid >> 5;
    const int e0   = blockIdx.x * MT;

    if (tid < MT) {
        int e = e0 + tid;
        int r = row[e], c = col[e];
        rowsh[tid] = r; colsh[tid] = c;
        atomicAdd(&cnt[r], 1.0f);
    }
    __syncthreads();

    for (int idx = tid; idx < MT * 32; idx += 256) {
        int m  = idx >> 5;
        int cp = idx & 31;
        const float4* src;
        int f;
        if (cp < 16) { src = (const float4*)(xb + (size_t)rowsh[m] * HH) + cp;        f = cp * 8; }
        else         { src = (const float4*)(xb + (size_t)colsh[m] * HH) + (cp - 16); f = HH + (cp - 16) * 8; }
        *(float4*)(A + m * S1 + f) = *src;
    }
    // attr + zero-pad tail: 32 values per edge (6 real, 26 pad).
    for (int idx = tid; idx < MT * 32; idx += 256) {
        int m = idx >> 5;
        int j = idx & 31;
        __bf16 v = (j < BDD) ? (__bf16)attr[(size_t)(e0 + m) * BDD + j] : (__bf16)0.0f;
        A[m * S1 + 2 * HH + j] = v;
    }
    __syncthreads();

    const int nbase = wave * 16;
    const int ncol  = nbase + (lane & 15);
    const int mofs  = (lane >> 4) << 3;

    v8f acc1[4] = {};
#pragma unroll
    for (int kt = 0; kt < K1 / 32; ++kt) {
        v16bf b  = frag_b_glb(W1T, K1, kt * 32, nbase, lane);
        v16bf a0 = frag_a_lds(A + 0 * 16 * S1, S1, kt * 32, lane);
        v16bf a1 = frag_a_lds(A + 1 * 16 * S1, S1, kt * 32, lane);
        v16bf a2 = frag_a_lds(A + 2 * 16 * S1, S1, kt * 32, lane);
        v16bf a3 = frag_a_lds(A + 3 * 16 * S1, S1, kt * 32, lane);
        acc1[0] = wmma_bf16(a0, b, acc1[0]);
        acc1[1] = wmma_bf16(a1, b, acc1[1]);
        acc1[2] = wmma_bf16(a2, b, acc1[2]);
        acc1[3] = wmma_bf16(a3, b, acc1[3]);
    }

    float bias1 = b1[ncol];
#pragma unroll
    for (int mt = 0; mt < 4; ++mt)
#pragma unroll
        for (int r = 0; r < 8; ++r) {
            float v = silu_f(acc1[mt][r] + bias1);
            H1[(mt * 16 + r + mofs) * S2 + ncol] = (__bf16)v;
        }
    __syncthreads();

    v8f acc2[4] = {};
#pragma unroll
    for (int kt = 0; kt < HH / 32; ++kt) {
        v16bf b  = frag_b_glb(W2T, HH, kt * 32, nbase, lane);
        v16bf a0 = frag_a_lds(H1 + 0 * 16 * S2, S2, kt * 32, lane);
        v16bf a1 = frag_a_lds(H1 + 1 * 16 * S2, S2, kt * 32, lane);
        v16bf a2 = frag_a_lds(H1 + 2 * 16 * S2, S2, kt * 32, lane);
        v16bf a3 = frag_a_lds(H1 + 3 * 16 * S2, S2, kt * 32, lane);
        acc2[0] = wmma_bf16(a0, b, acc2[0]);
        acc2[1] = wmma_bf16(a1, b, acc2[1]);
        acc2[2] = wmma_bf16(a2, b, acc2[2]);
        acc2[3] = wmma_bf16(a3, b, acc2[3]);
    }

    float bias2 = b2[ncol];
#pragma unroll
    for (int mt = 0; mt < 4; ++mt)
#pragma unroll
        for (int r = 0; r < 8; ++r) {
            int m = mt * 16 + r + mofs;
            atomicAdd(&acc[(size_t)rowsh[m] * HH + ncol], acc2[mt][r] + bias2);
        }
}

// ---------------------------------------------------------------------------
// Node kernel: 32 nodes/block (2 M-tiles/wave). attn MLP (768->128 via WMMA,
// 128->6 via VALU), softmax, scale blending, update MLP (WMMA), residual.
// ---------------------------------------------------------------------------
__launch_bounds__(256, 2)
__global__ void node_update(const float*  __restrict__ x,
                            const __bf16* __restrict__ xb,
                            const float*  __restrict__ acc,   // [6][N][128] scale means
                            const __bf16* __restrict__ attnW1T, const float* __restrict__ attn_b1,
                            const float*  __restrict__ attnW2,  const float* __restrict__ attn_b2,
                            const __bf16* __restrict__ updW1T,  const float* __restrict__ upd_b1,
                            const __bf16* __restrict__ updW2T,  const float* __restrict__ upd_b2,
                            float* __restrict__ out) {
    constexpr int MT = 32;                 // nodes per block
    constexpr int KA = HH * NSC;           // 768
    constexpr int SA = KA + 8;             // 776 (1552 B row, 16B-aligned)
    constexpr int SH = HH + 8;             // 136
    constexpr int KU = 2 * HH;             // 256
    constexpr int SU = KU + 8;             // 264
    __shared__ __bf16 Acat[MT * SA];
    __shared__ __bf16 Hh  [MT * SH];
    __shared__ __bf16 U   [MT * SU];
    __shared__ float  attw[MT][8];

    const int tid  = threadIdx.x;
    const int lane = tid & 31;
    const int wave = tid >> 5;
    const int n0   = blockIdx.x * MT;

    // Stage concat of the 6 scale outputs (scale-major blocks of 128), 4-wide.
    for (int idx = tid; idx < MT * (KA / 4); idx += 256) {
        int m = idx / (KA / 4);
        int q = idx - m * (KA / 4);
        int f = q * 4;
        int s = f >> 7, h = f & 127;
        float4 v = *(const float4*)&acc[((size_t)s * NND + (n0 + m)) * HH + h];
        __bf16* d = Acat + m * SA + f;
        d[0] = (__bf16)v.x; d[1] = (__bf16)v.y; d[2] = (__bf16)v.z; d[3] = (__bf16)v.w;
    }
    __syncthreads();

    const int nbase = wave * 16;
    const int ncol  = nbase + (lane & 15);
    const int mofs  = (lane >> 4) << 3;

    // attn hidden: [32,768] x [768,128]
    v8f a1[2] = {};
#pragma unroll
    for (int kt = 0; kt < KA / 32; ++kt) {
        v16bf b  = frag_b_glb(attnW1T, KA, kt * 32, nbase, lane);
        v16bf f0 = frag_a_lds(Acat + 0 * 16 * SA, SA, kt * 32, lane);
        v16bf f1 = frag_a_lds(Acat + 1 * 16 * SA, SA, kt * 32, lane);
        a1[0] = wmma_bf16(f0, b, a1[0]);
        a1[1] = wmma_bf16(f1, b, a1[1]);
    }
    {
        float bias = attn_b1[ncol];
#pragma unroll
        for (int mt = 0; mt < 2; ++mt)
#pragma unroll
            for (int r = 0; r < 8; ++r) {
                float v = silu_f(a1[mt][r] + bias);
                Hh[(mt * 16 + r + mofs) * SH + ncol] = (__bf16)v;
            }
    }
    __syncthreads();

    // logits [32,6] on VALU (tiny GEMM, N=6)
    if (tid < MT * NSC) {
        int m = tid / NSC, s = tid - m * NSC;
        float sum = attn_b2[s];
        for (int k = 0; k < HH; ++k)
            sum += (float)Hh[m * SH + k] * attnW2[(size_t)k * NSC + s];
        attw[m][s] = sum;
    }
    __syncthreads();
    if (tid < MT) {
        float mx = attw[tid][0];
        for (int s = 1; s < NSC; ++s) mx = fmaxf(mx, attw[tid][s]);
        float e[NSC], sum = 0.0f;
        for (int s = 0; s < NSC; ++s) { e[s] = __expf(attw[tid][s] - mx); sum += e[s]; }
        for (int s = 0; s < NSC; ++s) attw[tid][s] = e[s] / sum;
    }
    __syncthreads();

    // Build upd input U = [x | sum_s attn_s * scale_s]
    for (int idx = tid; idx < MT * HH; idx += 256) {
        int m = idx >> 7, h = idx & 127;
        U[m * SU + h] = xb[(size_t)(n0 + m) * HH + h];
        float wsum = 0.0f;
#pragma unroll
        for (int s = 0; s < NSC; ++s)
            wsum += (float)Acat[m * SA + s * HH + h] * attw[m][s];
        U[m * SU + HH + h] = (__bf16)wsum;
    }
    __syncthreads();

    // upd hidden: [32,256] x [256,128]
    v8f a2[2] = {};
#pragma unroll
    for (int kt = 0; kt < KU / 32; ++kt) {
        v16bf b  = frag_b_glb(updW1T, KU, kt * 32, nbase, lane);
        v16bf f0 = frag_a_lds(U + 0 * 16 * SU, SU, kt * 32, lane);
        v16bf f1 = frag_a_lds(U + 1 * 16 * SU, SU, kt * 32, lane);
        a2[0] = wmma_bf16(f0, b, a2[0]);
        a2[1] = wmma_bf16(f1, b, a2[1]);
    }
    {
        float bias = upd_b1[ncol];
#pragma unroll
        for (int mt = 0; mt < 2; ++mt)
#pragma unroll
            for (int r = 0; r < 8; ++r) {
                float v = silu_f(a2[mt][r] + bias);
                Hh[(mt * 16 + r + mofs) * SH + ncol] = (__bf16)v;
            }
    }
    __syncthreads();

    // upd out: [32,128] x [128,128] + bias + residual
    v8f a3[2] = {};
#pragma unroll
    for (int kt = 0; kt < HH / 32; ++kt) {
        v16bf b  = frag_b_glb(updW2T, HH, kt * 32, nbase, lane);
        v16bf f0 = frag_a_lds(Hh + 0 * 16 * SH, SH, kt * 32, lane);
        v16bf f1 = frag_a_lds(Hh + 1 * 16 * SH, SH, kt * 32, lane);
        a3[0] = wmma_bf16(f0, b, a3[0]);
        a3[1] = wmma_bf16(f1, b, a3[1]);
    }
    {
        float bias = upd_b2[ncol];
#pragma unroll
        for (int mt = 0; mt < 2; ++mt)
#pragma unroll
            for (int r = 0; r < 8; ++r) {
                int m = mt * 16 + r + mofs;
                size_t o = (size_t)(n0 + m) * HH + ncol;
                out[o] = x[o] + a3[mt][r] + bias;
            }
    }
}

// ---------------------------------------------------------------------------
// Host launcher
// ---------------------------------------------------------------------------
extern "C" void kernel_launch(void* const* d_in, const int* in_sizes, int n_in,
                              void* d_out, int out_size, void* d_ws, size_t ws_size,
                              hipStream_t stream) {
    const float* x      = (const float*)d_in[0];
    const float* pos    = (const float*)d_in[1];
    const int*   eidx   = (const int*)  d_in[2];   // [5][2][E]
    const int*   bidx   = (const int*)  d_in[3];   // [2][EB]
    const float* battr  = (const float*)d_in[4];   // [EB][6]
    const float* msgW1  = (const float*)d_in[5];   // [5][288][128]
    const float* msgb1  = (const float*)d_in[6];   // [5][128]
    const float* msgW2  = (const float*)d_in[7];   // [5][128][128]
    const float* msgb2  = (const float*)d_in[8];
    const float* bondW1 = (const float*)d_in[9];   // [262][128]
    const float* bondb1 = (const float*)d_in[10];
    const float* bondW2 = (const float*)d_in[11];  // [128][128]
    const float* bondb2 = (const float*)d_in[12];
    const float* attnW1 = (const float*)d_in[13];  // [768][128]
    const float* attnb1 = (const float*)d_in[14];
    const float* attnW2 = (const float*)d_in[15];  // [128][6]
    const float* attnb2 = (const float*)d_in[16];
    const float* updW1  = (const float*)d_in[17];  // [256][128]
    const float* updb1  = (const float*)d_in[18];
    const float* updW2  = (const float*)d_in[19];  // [128][128]
    const float* updb2  = (const float*)d_in[20];

    char*  ws  = (char*)d_ws;
    size_t off = 0;
    auto carve = [&](size_t bytes) -> void* {
        void* p = ws + off;
        off = (off + bytes + 255) & ~(size_t)255;
        return p;
    };

    __bf16* xb       = (__bf16*)carve((size_t)NND * HH * 2);
    __bf16* msgW1T   = (__bf16*)carve((size_t)5 * HH * 288 * 2);
    __bf16* msgW2T   = (__bf16*)carve((size_t)5 * HH * HH * 2);
    __bf16* bondW1T  = (__bf16*)carve((size_t)HH * 288 * 2);
    __bf16* bondW2T  = (__bf16*)carve((size_t)HH * HH * 2);
    __bf16* attnW1T  = (__bf16*)carve((size_t)HH * 768 * 2);
    __bf16* updW1T   = (__bf16*)carve((size_t)HH * 256 * 2);
    __bf16* updW2T   = (__bf16*)carve((size_t)HH * HH * 2);
    float*  acc      = (float*) carve((size_t)NSC * NND * HH * 4);
    float*  cnt      = (float*) carve((size_t)NSC * NND * 4);

    hipMemsetAsync(acc, 0, (size_t)NSC * NND * HH * 4, stream);
    hipMemsetAsync(cnt, 0, (size_t)NSC * NND * 4, stream);

    convert_x_kernel<<<(NND * HH + 255) / 256, 256, 0, stream>>>(x, xb, NND * HH);

    auto tconv = [&](const float* s, __bf16* d, int K, int Nc, int Kpad) {
        int total = Nc * Kpad;
        transpose_convert_kernel<<<(total + 255) / 256, 256, 0, stream>>>(s, d, K, Nc, Kpad);
    };
    for (int i = 0; i < 5; ++i) {
        tconv(msgW1 + (size_t)i * 288 * HH, msgW1T + (size_t)i * HH * 288, 288, HH, 288);
        tconv(msgW2 + (size_t)i * HH * HH,  msgW2T + (size_t)i * HH * HH,  HH,  HH, HH);
    }
    tconv(bondW1, bondW1T, 262, HH, 288);
    tconv(bondW2, bondW2T, HH,  HH, HH);
    tconv(attnW1, attnW1T, 768, HH, 768);
    tconv(updW1,  updW1T,  256, HH, 256);
    tconv(updW2,  updW2T,  HH,  HH, HH);

    const float CUT[5] = {5.0f, 10.0f, 15.0f, 20.0f, 25.0f};
    for (int i = 0; i < 5; ++i) {
        edge_mlp_spatial<<<EE / 64, 256, 0, stream>>>(
            xb, pos,
            eidx + (size_t)(i * 2 + 0) * EE,
            eidx + (size_t)(i * 2 + 1) * EE,
            msgW1T + (size_t)i * HH * 288, msgb1 + (size_t)i * HH,
            msgW2T + (size_t)i * HH * HH,  msgb2 + (size_t)i * HH,
            CUT[i],
            acc + (size_t)i * NND * HH, cnt + (size_t)i * NND);
    }
    edge_mlp_bond<<<EBB / 64, 256, 0, stream>>>(
        xb, bidx, bidx + EBB, battr,
        bondW1T, bondb1, bondW2T, bondb2,
        acc + (size_t)5 * NND * HH, cnt + (size_t)5 * NND);

    finalize_mean_kernel<<<(NSC * NND * HH + 255) / 256, 256, 0, stream>>>(
        acc, cnt, NSC * NND * HH);

    node_update<<<NND / 32, 256, 0, stream>>>(
        x, xb, acc,
        attnW1T, attnb1, attnW2, attnb2,
        updW1T, updb1, updW2T, updb2,
        (float*)d_out);
}
